// TimeAttn_28535762715240
// MI455X (gfx1250) — matrix operand order repaired and checked by reading, not compiled, not run on hardware
//
#include <hip/hip_runtime.h>
#include <hip/hip_bf16.h>

typedef __attribute__((ext_vector_type(16))) _Float16 v16h;
typedef __attribute__((ext_vector_type(8)))  _Float16 v8h;
typedef __attribute__((ext_vector_type(8)))  float    v8f;

#define B_  32
#define T_  12
#define N_  300
#define D_  128
#define M_  288
#define RSQRT_D 0.08838834764831845f   // 1/sqrt(128)

// ---------------------------------------------------------------------------
// Prep kernel: build half-precision WMMA-B-friendly operands in workspace.
//   wqh[n][k] = (half) wq[n][k]                               (128x128)
//   kh [m][d] = (half)((memory[m]·wk[d] + bk[d]) * 1/sqrt(D)) (288x128)
//   vT [j][m] = (half)( memory[m]·wv[j] + bv[j])              (128x288)
// ---------------------------------------------------------------------------
__global__ __launch_bounds__(256) void prep_kernel(
    const float* __restrict__ memory,
    const float* __restrict__ wq,
    const float* __restrict__ wk, const float* __restrict__ bk,
    const float* __restrict__ wv, const float* __restrict__ bv,
    _Float16* __restrict__ wqh, _Float16* __restrict__ kh,
    _Float16* __restrict__ vT)
{
    const int idx = blockIdx.x * 256 + threadIdx.x;
    if (blockIdx.y == 0) {
        if (idx < D_ * D_) wqh[idx] = (_Float16)wq[idx];
    } else if (blockIdx.y == 1) {
        if (idx < M_ * D_) {
            const int m = idx / D_, j = idx % D_;
            const float4* mr = (const float4*)(memory + (size_t)m * D_);
            const float4* wr = (const float4*)(wk + (size_t)j * D_);
            float acc = bk[j];
            #pragma unroll 8
            for (int d = 0; d < D_ / 4; ++d) {
                const float4 a = mr[d], w = wr[d];
                acc += a.x * w.x + a.y * w.y + a.z * w.z + a.w * w.w;
            }
            kh[idx] = (_Float16)(acc * RSQRT_D);
        }
    } else {
        if (idx < D_ * M_) {
            const int j = idx / M_, m = idx % M_;
            const float4* mr = (const float4*)(memory + (size_t)m * D_);
            const float4* wr = (const float4*)(wv + (size_t)j * D_);
            float acc = bv[j];
            #pragma unroll 8
            for (int d = 0; d < D_ / 4; ++d) {
                const float4 a = mr[d], w = wr[d];
                acc += a.x * w.x + a.y * w.y + a.z * w.z + a.w * w.w;
            }
            vT[idx] = (_Float16)acc;
        }
    }
}

// ---------------------------------------------------------------------------
// Fragment loaders (wave32, 16x16x32 f16 WMMA; this file's contiguous k order, same for A and B)
//  A (16x32): lane L -> row = L&15, g = L>>4;
//             frag[0..7]  = A[row][k0 + g*16 + i]
//             frag[8..15] = A[row][k0 + g*16 + 8 + i]
//  B (32x16): lane L -> col = L&15, g = L>>4; frag[i] = B[k0 + g*16 + i][col]
//             (stored pre-transposed so the 16 halfs are contiguous)
// ---------------------------------------------------------------------------
__device__ __forceinline__ v16h loadA_h(const _Float16* buf, int stride,
                                        int k0, int row, int g)
{
    const v8h lo = *(const v8h*)(buf + row * stride + k0 + g * 16);
    const v8h hi = *(const v8h*)(buf + row * stride + k0 + g * 16 + 8);
    return __builtin_shufflevector(lo, hi, 0, 1, 2, 3, 4, 5, 6, 7,
                                           8, 9, 10, 11, 12, 13, 14, 15);
}

__device__ __forceinline__ v16h loadA_f(const float* buf, int stride,
                                        int k0, int row, int g)
{
    const float4 a0 = *(const float4*)(buf + row * stride + k0 + g * 16);
    const float4 a1 = *(const float4*)(buf + row * stride + k0 + g * 16 + 4);
    const float4 b0 = *(const float4*)(buf + row * stride + k0 + g * 16 + 8);
    const float4 b1 = *(const float4*)(buf + row * stride + k0 + g * 16 + 12);
    v16h r;
    r[0]  = (_Float16)a0.x; r[1]  = (_Float16)a0.y;
    r[2]  = (_Float16)a0.z; r[3]  = (_Float16)a0.w;
    r[4]  = (_Float16)a1.x; r[5]  = (_Float16)a1.y;
    r[6]  = (_Float16)a1.z; r[7]  = (_Float16)a1.w;
    r[8]  = (_Float16)b0.x; r[9]  = (_Float16)b0.y;
    r[10] = (_Float16)b0.z; r[11] = (_Float16)b0.w;
    r[12] = (_Float16)b1.x; r[13] = (_Float16)b1.y;
    r[14] = (_Float16)b1.z; r[15] = (_Float16)b1.w;
    return r;
}

__device__ __forceinline__ v8f wmma_f16(v16h a, v16h b, v8f c)
{
    return __builtin_amdgcn_wmma_f32_16x16x32_f16(
        /*neg_a=*/false, a, /*neg_b=*/false, b,
        /*c_mod=*/(short)0, c, /*reuse_a=*/false, /*reuse_b=*/false);
}

// ---------------------------------------------------------------------------
// Main fused attention kernel: one wave32 per (b, n); T=12 rows padded to 16.
// ---------------------------------------------------------------------------
__global__ __launch_bounds__(32) void attn_kernel(
    const float* __restrict__ x,
    const float* __restrict__ memory,
    const float* __restrict__ bq,
    const _Float16* __restrict__ wqh,
    const _Float16* __restrict__ kh,
    const _Float16* __restrict__ vT,
    float* __restrict__ out)
{
    __shared__ __align__(32) _Float16 sx[16 * D_];   // x tile, half
    __shared__ __align__(32) _Float16 sq[16 * D_];   // q tile, half
    __shared__ __align__(32) float    sp[16 * M_];   // scores -> probs, f32

    const int wg   = blockIdx.x;       // 0 .. B_*N_-1
    const int b    = wg / N_;
    const int n    = wg % N_;
    const int lane = threadIdx.x;      // 0..31
    const int row  = lane & 15;
    const int g    = lane >> 4;

    const size_t plane  = (size_t)B_ * T_ * N_ * D_;
    float* out0 = out;                  // attention output
    float* out1 = out + plane;          // x pass-through
    float* out2 = out + 2 * plane;      // pos = memory[top1]
    float* out3 = out + 3 * plane;      // neg = memory[top2]

    const size_t rstride = (size_t)N_ * D_;          // stride between t rows
    const float* xbase = x + ((size_t)b * T_ * N_ + n) * D_;

    // ---- stage x tile to LDS (half), emit exact f32 copy of x -------------
    for (int t = 0; t < T_; ++t) {
        const float4 f = *(const float4*)(xbase + t * rstride + lane * 4);
        *(float4*)(out1 + ((size_t)(b * T_ + t) * N_ + n) * D_ + lane * 4) = f;
        _Float16* d = &sx[t * D_ + lane * 4];
        d[0] = (_Float16)f.x; d[1] = (_Float16)f.y;
        d[2] = (_Float16)f.z; d[3] = (_Float16)f.w;
    }
    for (int t = T_; t < 16; ++t) {                  // zero pad rows
        _Float16* d = &sx[t * D_ + lane * 4];
        d[0] = d[1] = d[2] = d[3] = (_Float16)0.0f;
    }

    // ---- q = x @ wq^T + bq  ->  sq (half) ---------------------------------
    {
        // A fragments depend only on the k-slice: load the 4 slices once.
        v16h ax[4];
        #pragma unroll
        for (int k = 0; k < 4; ++k) ax[k] = loadA_h(sx, D_, k * 32, row, g);

        #pragma unroll
        for (int n0 = 0; n0 < D_ / 16; ++n0) {
            const float bias = bq[n0 * 16 + row];
            v8f acc;
            #pragma unroll
            for (int i = 0; i < 8; ++i) acc[i] = bias;
            #pragma unroll
            for (int k = 0; k < 4; ++k) {
                const v16h bf = *(const v16h*)(wqh
                    + (size_t)(n0 * 16 + row) * D_ + k * 32 + g * 16);
                acc = wmma_f16(ax[k], bf, acc);
            }
            #pragma unroll
            for (int v = 0; v < 8; ++v)
                sq[(v + g * 8) * D_ + n0 * 16 + row] = (_Float16)acc[v];
        }
    }

    // ---- scores = q @ k^T / sqrt(D)  ->  sp (f32) -------------------------
    {
        v16h aq[4];
        #pragma unroll
        for (int k = 0; k < 4; ++k) aq[k] = loadA_h(sq, D_, k * 32, row, g);

        #pragma unroll 3
        for (int m0 = 0; m0 < M_ / 16; ++m0) {
            v8f acc = {};
            #pragma unroll
            for (int k = 0; k < 4; ++k) {
                const v16h bf = *(const v16h*)(kh
                    + (size_t)(m0 * 16 + row) * D_ + k * 32 + g * 16);
                acc = wmma_f16(aq[k], bf, acc);
            }
            #pragma unroll
            for (int v = 0; v < 8; ++v)
                sp[(v + g * 8) * M_ + m0 * 16 + row] = acc[v];
        }
    }

    // ---- softmax (f32, full wave per row) ---------------------------------
    for (int t = 0; t < T_; ++t) {
        float vals[M_ / 32];
        float vmax = -3.0e38f;
        #pragma unroll
        for (int i = 0; i < M_ / 32; ++i) {
            vals[i] = sp[t * M_ + lane + 32 * i];
            vmax = fmaxf(vmax, vals[i]);
        }
        #pragma unroll
        for (int off = 16; off; off >>= 1)
            vmax = fmaxf(vmax, __shfl_xor(vmax, off, 32));
        float sum = 0.0f;
        #pragma unroll
        for (int i = 0; i < M_ / 32; ++i) {
            vals[i] = __expf(vals[i] - vmax);
            sum += vals[i];
        }
        #pragma unroll
        for (int off = 16; off; off >>= 1)
            sum += __shfl_xor(sum, off, 32);
        const float inv = 1.0f / sum;
        #pragma unroll
        for (int i = 0; i < M_ / 32; ++i)
            sp[t * M_ + lane + 32 * i] = vals[i] * inv;
    }

    // ---- top-2 per row + gather memory rows (pos/neg) ---------------------
    for (int t = 0; t < T_; ++t) {
        float v0 = -1.0f, v1 = -2.0f;
        int   i0 = 0,     i1 = 0;
        #pragma unroll
        for (int i = 0; i < M_ / 32; ++i) {
            const int   m = lane + 32 * i;
            const float p = sp[t * M_ + m];
            if (p > v0)      { v1 = v0; i1 = i0; v0 = p; i0 = m; }
            else if (p > v1) { v1 = p;  i1 = m; }
        }
        #pragma unroll
        for (int off = 16; off; off >>= 1) {
            const float ov0 = __shfl_xor(v0, off, 32);
            const int   oi0 = __shfl_xor(i0, off, 32);
            const float ov1 = __shfl_xor(v1, off, 32);
            const int   oi1 = __shfl_xor(i1, off, 32);
            if (ov0 > v0) {
                if (v0 > ov1) { v1 = v0;  i1 = i0;  }
                else          { v1 = ov1; i1 = oi1; }
                v0 = ov0; i0 = oi0;
            } else if (ov0 > v1) {
                v1 = ov0; i1 = oi0;
            }
        }
        const size_t drow = ((size_t)(b * T_ + t) * N_ + n) * D_;
        const float4 pr = *(const float4*)(memory + (size_t)i0 * D_ + lane * 4);
        *(float4*)(out2 + drow + lane * 4) = pr;
        const float4 nr = *(const float4*)(memory + (size_t)i1 * D_ + lane * 4);
        *(float4*)(out3 + drow + lane * 4) = nr;
    }

    // ---- out = attn @ v ----------------------------------------------------
    {
        // Cache the 9 attn A-fragments (f32 -> f16 convert done once each).
        v16h ap[M_ / 32];
        #pragma unroll
        for (int k = 0; k < M_ / 32; ++k)
            ap[k] = loadA_f(sp, M_, k * 32, row, g);

        #pragma unroll
        for (int n0 = 0; n0 < D_ / 16; ++n0) {
            v8f acc = {};
            #pragma unroll
            for (int k = 0; k < M_ / 32; ++k) {
                const v16h bf = *(const v16h*)(vT
                    + (size_t)(n0 * 16 + row) * M_ + k * 32 + g * 16);
                acc = wmma_f16(ap[k], bf, acc);
            }
            #pragma unroll
            for (int v = 0; v < 8; ++v) {
                const int t = v + g * 8;
                if (t < T_)
                    out0[((size_t)(b * T_ + t) * N_ + n) * D_ + n0 * 16 + row]
                        = acc[v];
            }
        }
    }
}

// ---------------------------------------------------------------------------
extern "C" void kernel_launch(void* const* d_in, const int* in_sizes, int n_in,
                              void* d_out, int out_size, void* d_ws, size_t ws_size,
                              hipStream_t stream)
{
    const float* x      = (const float*)d_in[0];
    const float* memory = (const float*)d_in[1];
    const float* wq     = (const float*)d_in[2];
    const float* bq     = (const float*)d_in[3];
    const float* wk     = (const float*)d_in[4];
    const float* bk     = (const float*)d_in[5];
    const float* wv     = (const float*)d_in[6];
    const float* bv     = (const float*)d_in[7];
    float* out = (float*)d_out;

    // Workspace layout (all offsets 256B-aligned)
    char* ws = (char*)d_ws;
    _Float16* wqh = (_Float16*)(ws);                     // 128*128*2 = 32768 B
    _Float16* kh  = (_Float16*)(ws + 32768);             // 288*128*2 = 73728 B
    _Float16* vT  = (_Float16*)(ws + 32768 + 73728);     // 128*288*2 = 73728 B

    // 1) Build half-precision operands (K, V, wq) once.
    dim3 pgrid((M_ * D_ + 255) / 256, 3);
    prep_kernel<<<pgrid, 256, 0, stream>>>(memory, wq, wk, bk, wv, bv,
                                           wqh, kh, vT);

    // 2) Fused attention: one wave32 per (batch, sensor).
    attn_kernel<<<B_ * N_, 32, 0, stream>>>(x, memory, bq, wqh, kh, vT, out);
}
